// SchNetInteractionBlock_45037027066141
// MI455X (gfx1250) — compile-verified, matrix operand order checked by reading, not run on hardware
//
#include <hip/hip_runtime.h>
#include <hip/hip_bf16.h>

// ---------------------------------------------------------------------------
// SchNet interaction block for gfx1250 (CDNA5), bf16 WMMA path.
//   h1    = ssp(dijk @ W1 + b1)            [n_edges,128]
//   w_ijk = ssp(h1 @ W2 + b2)              [n_edges,128]
//   (fused) agg[seg_i[seg_j[e]]] += xf[idx_j[seg_j[e]]] * w_ijk[e]
//   xf    = x @ Wi
//   h     = ssp(agg @ Wo + bo); v = h @ Wd + bd; y = x + v
// All GEMMs: v_wmma_f32_16x16x32_bf16, f32 accumulate.
// ---------------------------------------------------------------------------

#define NF 128  // n_basis == n_filters == n_in == 128

typedef __bf16  bf16_t;
typedef bf16_t  v16bf __attribute__((ext_vector_type(16)));
typedef float   v8f   __attribute__((ext_vector_type(8)));

__device__ __forceinline__ unsigned short f2bfu(float f) {
  union { float f; unsigned u; } x; x.f = f;
  unsigned r = x.u + 0x7FFFu + ((x.u >> 16) & 1u);   // round-to-nearest-even
  return (unsigned short)(r >> 16);
}

__device__ __forceinline__ float sspf(float x) {
  // shifted softplus: log(0.5*exp(x)+0.5) = max(x,0)+log1p(exp(-|x|))-log2
  return fmaxf(x, 0.0f) + __logf(1.0f + __expf(-fabsf(x))) - 0.69314718055994531f;
}

// Load one 16x32 bf16 WMMA operand fragment from a row-major [rows][128] bf16
// LDS tile. Lane layout (A and B identical, B tile stored as W^T[n][k]):
//   lane L: row = L&15, half = L>>4
//   vector elems 0..7  = [row][k0 + 0..7]      (k0 includes half*8)
//   vector elems 8..15 = [row][k0 + 16 .. 23]
__device__ __forceinline__ v16bf ldfrag(const unsigned short* base, int row, int k0) {
  union { v16bf v; uint4 q[2]; } u;
  const uint4* p = (const uint4*)(base + row * NF + k0);
  u.q[0] = p[0];      // 8 bf16
  u.q[1] = p[2];      // +16 elements
  return u.v;
}

// A-fragments for the wave's 16x128 tile are invariant across N-tiles: load
// the 4 K-step fragments (32 VGPRs) once and reuse for all 8 N-tiles.
__device__ __forceinline__ void load_afrags(const unsigned short* sA, int lane,
                                            v16bf a[4]) {
  const int nloc = lane & 15;
  const int half = lane >> 4;
#pragma unroll
  for (int ks = 0; ks < 4; ++ks) a[ks] = ldfrag(sA, nloc, ks * 32 + half * 8);
}

// One 16(M) x 16(N) tile: K reduced in 4 steps of 32.
__device__ __forceinline__ v8f tile_gemm(const v16bf a[4],
                                         const unsigned short* sBt,
                                         int nT, int lane) {
  const int nloc = lane & 15;
  const int half = lane >> 4;
  v8f acc = {};
#pragma unroll
  for (int ks = 0; ks < 4; ++ks) {
    v16bf b = ldfrag(sBt, nT * 16 + nloc, ks * 32 + half * 8);
    acc = __builtin_amdgcn_wmma_f32_16x16x32_bf16(
        /*neg_a=*/false, a[ks], /*neg_b=*/false, b,
        /*c_mod=*/(short)0, acc, /*reuse_a=*/false, /*reuse_b=*/false);
  }
  return acc;
}

__device__ __forceinline__ void atomic_add_f32(float* p, float v) {
  (void)__hip_atomic_fetch_add(p, v, __ATOMIC_RELAXED, __HIP_MEMORY_SCOPE_AGENT);
}

// ---------------------------------------------------------------------------
// Weight transpose + f32->bf16: dst[n*128+k] = bf16(src[k*128+n])
// ---------------------------------------------------------------------------
__global__ __launch_bounds__(NF) void wtrans_kernel(const float* __restrict__ src,
                                                    unsigned short* __restrict__ dst) {
  const int n = blockIdx.x;
  const int k = threadIdx.x;
  dst[n * NF + k] = f2bfu(src[k * NF + n]);
}

// ---------------------------------------------------------------------------
// xf = x @ Wi   (rows = n_atoms; 4 waves/block, 16 rows/wave)
// ---------------------------------------------------------------------------
__global__ __launch_bounds__(128) void xf_kernel(const float* __restrict__ x,
                                                 const unsigned short* __restrict__ Wit,
                                                 float* __restrict__ xf,
                                                 int n_rows) {
  __shared__ __align__(16) unsigned short sW[NF * NF];
  __shared__ __align__(16) unsigned short sA[4][16 * NF];

  const int tid  = threadIdx.x;
  const int wave = tid >> 5;
  const int lane = tid & 31;

  {  // cooperative weight load (32 KB)
    const unsigned* g = (const unsigned*)Wit;
    unsigned* s = (unsigned*)sW;
    for (int i = tid; i < NF * NF / 2; i += 128) s[i] = g[i];
  }

  const int rBase = blockIdx.x * 64 + wave * 16;
  for (int t = lane; t < 512; t += 32) {  // 16 rows x 128 cols as float4
    int r = t >> 5, c4 = (t & 31) << 2;
    int row = rBase + r; if (row > n_rows - 1) row = n_rows - 1;
    float4 f = *(const float4*)(x + (size_t)row * NF + c4);
    unsigned short* d = &sA[wave][r * NF + c4];
    d[0] = f2bfu(f.x); d[1] = f2bfu(f.y); d[2] = f2bfu(f.z); d[3] = f2bfu(f.w);
  }
  __syncthreads();

  const int nloc  = lane & 15;
  const int mbase = (lane >> 4) * 8;
  const bool fullTile = (rBase + 16 <= n_rows);  // wave-uniform

  v16bf af[4];
  load_afrags(&sA[wave][0], lane, af);

#pragma unroll
  for (int nT = 0; nT < 8; ++nT) {
    v8f acc = tile_gemm(af, sW, nT, lane);
    const int col = nT * 16 + nloc;
    if (fullTile) {
      float* p = xf + (size_t)(rBase + mbase) * NF + col;
#pragma unroll
      for (int v = 0; v < 8; ++v) p[v * NF] = acc[v];
    } else {
#pragma unroll
      for (int v = 0; v < 8; ++v) {
        const int row = rBase + mbase + v;
        if (row < n_rows) xf[(size_t)row * NF + col] = acc[v];
      }
    }
  }
}

// ---------------------------------------------------------------------------
// Edge kernel: two dense+ssp layers on dijk, then fused gather/scale/scatter.
// 4 waves/block, 16 edges/wave, 64 edges/block. Scatter atomics are
// run-length merged: i = seg_i[seg_j[e]] is non-decreasing in e.
// ---------------------------------------------------------------------------
__global__ __launch_bounds__(128) void edge_kernel(const float* __restrict__ dijk,
                                                   const int* __restrict__ idx_j,
                                                   const int* __restrict__ seg_i,
                                                   const int* __restrict__ seg_j,
                                                   const float* __restrict__ b1,
                                                   const float* __restrict__ b2,
                                                   const unsigned short* __restrict__ W1t,
                                                   const unsigned short* __restrict__ W2t,
                                                   const float* __restrict__ xf,
                                                   float* __restrict__ agg,
                                                   int n_edges) {
  __shared__ __align__(16) unsigned short sW1[NF * NF];   // 32 KB
  __shared__ __align__(16) unsigned short sW2[NF * NF];   // 32 KB
  __shared__ __align__(16) unsigned short sA[4][16 * NF]; // 16 KB
  __shared__ __align__(16) unsigned short sH[4][16 * NF]; // 16 KB
  __shared__ int sJ[4][16];
  __shared__ int sI[4][16];

  const int tid  = threadIdx.x;
  const int wave = tid >> 5;
  const int lane = tid & 31;

  {  // cooperative load of both weight matrices
    const unsigned* g1 = (const unsigned*)W1t;
    const unsigned* g2 = (const unsigned*)W2t;
    unsigned* s1 = (unsigned*)sW1;
    unsigned* s2 = (unsigned*)sW2;
    for (int i = tid; i < NF * NF / 2; i += 128) { s1[i] = g1[i]; s2[i] = g2[i]; }
  }

  const int eBase = blockIdx.x * 64 + wave * 16;
  if (lane < 16) {  // fold seg_j -> (idx_j, seg_i) once per edge
    int e = eBase + lane; if (e > n_edges - 1) e = n_edges - 1;
    const int p = seg_j[e];
    sJ[wave][lane] = idx_j[p];
    sI[wave][lane] = seg_i[p];
  }
  for (int t = lane; t < 512; t += 32) {  // stage dijk tile as bf16
    int r = t >> 5, c4 = (t & 31) << 2;
    int e = eBase + r; if (e > n_edges - 1) e = n_edges - 1;
    float4 f = *(const float4*)(dijk + (size_t)e * NF + c4);
    unsigned short* d = &sA[wave][r * NF + c4];
    d[0] = f2bfu(f.x); d[1] = f2bfu(f.y); d[2] = f2bfu(f.z); d[3] = f2bfu(f.w);
  }
  __syncthreads();

  const int nloc  = lane & 15;
  const int mbase = (lane >> 4) * 8;

  // ----- layer 1: h1 = ssp(dijk @ W1 + b1) -> sH (bf16) -----
  {
    v16bf af[4];
    load_afrags(&sA[wave][0], lane, af);
#pragma unroll
    for (int nT = 0; nT < 8; ++nT) {
      v8f acc = tile_gemm(af, sW1, nT, lane);
      const int col = nT * 16 + nloc;
      const float bv = b1[col];
#pragma unroll
      for (int v = 0; v < 8; ++v)
        sH[wave][(mbase + v) * NF + col] = f2bfu(sspf(acc[v] + bv));
    }
  }
  __syncthreads();

  // preload this lane's 8 (i, j) pairs and validity (uniform per half-wave)
  int ii[8], jj[8];
  bool ok[8];
#pragma unroll
  for (int v = 0; v < 8; ++v) {
    ii[v] = sI[wave][mbase + v];
    jj[v] = sJ[wave][mbase + v];
    ok[v] = (eBase + mbase + v) < n_edges;
  }

  // ----- layer 2 + fused gather * run-length-merged scatter-add -----
  {
    v16bf af[4];
    load_afrags(&sH[wave][0], lane, af);
#pragma unroll
    for (int nT = 0; nT < 8; ++nT) {
      v8f acc = tile_gemm(af, sW2, nT, lane);
      const int col = nT * 16 + nloc;
      const float bv = b2[col];
      float pend  = 0.0f;
      int   ipend = ii[0];
#pragma unroll
      for (int v = 0; v < 8; ++v) {
        float val = sspf(acc[v] + bv) * xf[(size_t)jj[v] * NF + col];  // msg
        val = ok[v] ? val : 0.0f;                                      // cndmask
        if (ii[v] != ipend) {  // half-wave-uniform branch
          atomic_add_f32(&agg[(size_t)ipend * NF + col], pend);
          pend = 0.0f;
          ipend = ii[v];
        }
        pend += val;
      }
      atomic_add_f32(&agg[(size_t)ipend * NF + col], pend);
    }
  }
}

// ---------------------------------------------------------------------------
// Output kernel: h = ssp(agg@Wo+bo); v = h@Wd+bd; y = x+v.
// d_out layout: [y (n_atoms*128) | v (n_atoms*128)]
// ---------------------------------------------------------------------------
__global__ __launch_bounds__(128) void out_kernel(const float* __restrict__ agg,
                                                  const float* __restrict__ x,
                                                  const unsigned short* __restrict__ Wot,
                                                  const unsigned short* __restrict__ Wdt,
                                                  const float* __restrict__ bo,
                                                  const float* __restrict__ bd,
                                                  float* __restrict__ out,
                                                  int n_rows) {
  __shared__ __align__(16) unsigned short sWo[NF * NF];
  __shared__ __align__(16) unsigned short sWd[NF * NF];
  __shared__ __align__(16) unsigned short sA[4][16 * NF];
  __shared__ __align__(16) unsigned short sH[4][16 * NF];

  const int tid  = threadIdx.x;
  const int wave = tid >> 5;
  const int lane = tid & 31;

  {
    const unsigned* g1 = (const unsigned*)Wot;
    const unsigned* g2 = (const unsigned*)Wdt;
    unsigned* s1 = (unsigned*)sWo;
    unsigned* s2 = (unsigned*)sWd;
    for (int i = tid; i < NF * NF / 2; i += 128) { s1[i] = g1[i]; s2[i] = g2[i]; }
  }

  const int rBase = blockIdx.x * 64 + wave * 16;
  for (int t = lane; t < 512; t += 32) {
    int r = t >> 5, c4 = (t & 31) << 2;
    int row = rBase + r; if (row > n_rows - 1) row = n_rows - 1;
    float4 f = *(const float4*)(agg + (size_t)row * NF + c4);
    unsigned short* d = &sA[wave][r * NF + c4];
    d[0] = f2bfu(f.x); d[1] = f2bfu(f.y); d[2] = f2bfu(f.z); d[3] = f2bfu(f.w);
  }
  __syncthreads();

  const int nloc  = lane & 15;
  const int mbase = (lane >> 4) * 8;
  const bool fullTile = (rBase + 16 <= n_rows);  // wave-uniform

  // h = ssp(agg @ Wo + bo)
  {
    v16bf af[4];
    load_afrags(&sA[wave][0], lane, af);
#pragma unroll
    for (int nT = 0; nT < 8; ++nT) {
      v8f acc = tile_gemm(af, sWo, nT, lane);
      const int col = nT * 16 + nloc;
      const float bv = bo[col];
#pragma unroll
      for (int v = 0; v < 8; ++v)
        sH[wave][(mbase + v) * NF + col] = f2bfu(sspf(acc[v] + bv));
    }
  }
  __syncthreads();

  // v = h @ Wd + bd ; y = x + v
  const size_t vOff = (size_t)n_rows * NF;
  {
    v16bf af[4];
    load_afrags(&sH[wave][0], lane, af);
#pragma unroll
    for (int nT = 0; nT < 8; ++nT) {
      v8f acc = tile_gemm(af, sWd, nT, lane);
      const int col = nT * 16 + nloc;
      const float bv = bd[col];
      if (fullTile) {
        const size_t o0 = (size_t)(rBase + mbase) * NF + col;
        const float* px = x + o0;
        float* py = out + o0;
        float* pv = out + vOff + o0;
#pragma unroll
        for (int v = 0; v < 8; ++v) {
          const float vv = acc[v] + bv;
          pv[v * NF] = vv;              // v
          py[v * NF] = px[v * NF] + vv; // y = x + v
        }
      } else {
#pragma unroll
        for (int v = 0; v < 8; ++v) {
          const int row = rBase + mbase + v;
          if (row < n_rows) {
            const float vv = acc[v] + bv;
            const size_t o = (size_t)row * NF + col;
            out[vOff + o] = vv;
            out[o] = x[o] + vv;
          }
        }
      }
    }
  }
}

// ---------------------------------------------------------------------------
// Host launcher (graph-capture safe: only async ops on `stream`)
// ---------------------------------------------------------------------------
extern "C" void kernel_launch(void* const* d_in, const int* in_sizes, int n_in,
                              void* d_out, int out_size, void* d_ws, size_t ws_size,
                              hipStream_t stream) {
  const float* x    = (const float*)d_in[0];
  const float* dijk = (const float*)d_in[1];
  const int*   idx_j = (const int*)d_in[2];
  const int*   seg_i = (const int*)d_in[3];
  const int*   seg_j = (const int*)d_in[4];
  const float* W1 = (const float*)d_in[5];
  const float* b1 = (const float*)d_in[6];
  const float* W2 = (const float*)d_in[7];
  const float* b2 = (const float*)d_in[8];
  const float* Wi = (const float*)d_in[9];
  const float* Wo = (const float*)d_in[10];
  const float* bo = (const float*)d_in[11];
  const float* Wd = (const float*)d_in[12];
  const float* bd = (const float*)d_in[13];

  const int n_atoms = in_sizes[0] / NF;
  const int n_edges = in_sizes[2];

  // workspace carve-up
  char* w = (char*)d_ws;
  float* agg = (float*)w; w += (size_t)n_atoms * NF * sizeof(float);
  float* xf  = (float*)w; w += (size_t)n_atoms * NF * sizeof(float);
  unsigned short* W1t = (unsigned short*)w; w += NF * NF * sizeof(unsigned short);
  unsigned short* W2t = (unsigned short*)w; w += NF * NF * sizeof(unsigned short);
  unsigned short* Wit = (unsigned short*)w; w += NF * NF * sizeof(unsigned short);
  unsigned short* Wot = (unsigned short*)w; w += NF * NF * sizeof(unsigned short);
  unsigned short* Wdt = (unsigned short*)w; w += NF * NF * sizeof(unsigned short);

  hipMemsetAsync(agg, 0, (size_t)n_atoms * NF * sizeof(float), stream);

  wtrans_kernel<<<NF, NF, 0, stream>>>(W1, W1t);
  wtrans_kernel<<<NF, NF, 0, stream>>>(W2, W2t);
  wtrans_kernel<<<NF, NF, 0, stream>>>(Wi, Wit);
  wtrans_kernel<<<NF, NF, 0, stream>>>(Wo, Wot);
  wtrans_kernel<<<NF, NF, 0, stream>>>(Wd, Wdt);

  const int ablocks = (n_atoms + 63) / 64;
  const int eblocks = (n_edges + 63) / 64;

  xf_kernel<<<ablocks, 128, 0, stream>>>(x, Wit, xf, n_atoms);
  edge_kernel<<<eblocks, 128, 0, stream>>>(dijk, idx_j, seg_i, seg_j, b1, b2,
                                           W1t, W2t, xf, agg, n_edges);
  out_kernel<<<ablocks, 128, 0, stream>>>(agg, x, Wot, Wdt, bo, bd,
                                          (float*)d_out, n_atoms);
}
